// FullAttentionBlock_53317724013007
// MI455X (gfx1250) — matrix operand order repaired and checked by reading, not compiled, not run on hardware
//
#include <hip/hip_runtime.h>
#include <hip/hip_bf16.h>

// ---------------- problem dims (from reference) ----------------
#define DIMD 2048
#define NH   32
#define NKV  4
#define HD   128
#define ROT  64
#define NE   8
#define MIH  1408
#define SIH  5632
#define BB   2
#define SSQ  512
#define MCC  1024
#define TT   (BB*SSQ)
#define EPSF 1e-6f

typedef __bf16 bf16_t;
typedef __attribute__((ext_vector_type(16))) __bf16 v16bf;
typedef __attribute__((ext_vector_type(8)))  float  v8f;

__device__ __forceinline__ bf16_t f2bf(float f) { return (bf16_t)f; }

// pack 4 f32 -> 4 bf16, single 64-bit LDS store (dst must be 8B aligned)
__device__ __forceinline__ void store_bf16x4(bf16_t* dst, float4 f) {
  union { bf16_t b[4]; uint2 u; } t;
  t.b[0] = (bf16_t)f.x; t.b[1] = (bf16_t)f.y;
  t.b[2] = (bf16_t)f.z; t.b[3] = (bf16_t)f.w;
  *(uint2*)dst = t.u;
}

struct alignas(16) BF8 { bf16_t v[8]; };

// A-fragment (16x32 bf16): lane 0-15 row=lane, K: elems0-7 = half*8+0..7,
// elems8-15 = 16+half*8+0..7  (half = lane>>4)
__device__ __forceinline__ v16bf load_a_frag(const bf16_t* row, int half) {
  union { v16bf v; BF8 h[2]; } u;
  u.h[0] = *(const BF8*)(row + half * 8);
  u.h[1] = *(const BF8*)(row + 16 + half * 8);
  return u.v;
}
// B-fragment (32x16 bf16, stored as B^T rows w/ K contiguous):
// lane n=lane&15 holds K: elems0-7 = half*8+0..7, elems8-15 = 16+half*8+0..7
__device__ __forceinline__ v16bf load_b_frag(const bf16_t* row, int half) {
  union { v16bf v; BF8 h[2]; } u;
  u.h[0] = *(const BF8*)(row + half * 8);
  u.h[1] = *(const BF8*)(row + 16 + half * 8);
  return u.v;
}
__device__ __forceinline__ v8f wmma_bf16(v16bf a, v16bf b, v8f c) {
  return __builtin_amdgcn_wmma_f32_16x16x32_bf16(false, a, false, b, (short)0, c,
                                                 false, false);
}
__device__ __forceinline__ float sigmoidf_(float x) {
  return 1.0f / (1.0f + __expf(-x));
}

// ================== generic tiled WMMA GEMM ==================
// C[M,N] = A[M,K] @ W + epilogue.  W layout: wTrans=0 -> (K,N), wTrans=1 -> (N,K).
// epilogue: v = acc * rowScale[m*rsStride]? + residual[m,n]? + C[m,n] (if accum)
// All M % 64 == 0, N % 128 == 0, K % 64 == 0 in this problem.
// Software-pipelined: next tile's global loads issued before compute so HBM
// latency overlaps the WMMA phase; staging is two-phase (batch loads, one wait).
#define BM 64
#define BN 128
#define BK 64
__global__ __launch_bounds__(256)
void gemm_wmma(const float* __restrict__ A, const float* __restrict__ W,
               float* __restrict__ C, int M, int N, int K, int wTrans,
               const float* __restrict__ residual,
               const float* __restrict__ rowScale, int rsStride, int accumulate) {
  __shared__ __attribute__((aligned(16))) bf16_t lA[BM][BK];
  __shared__ __attribute__((aligned(16))) bf16_t lB[BN][BK];
  const int tid = threadIdx.x;
  const int lane = tid & 31, wid = tid >> 5;
  const int half = lane >> 4, l16 = lane & 15;
  const int m0 = blockIdx.y * BM;
  const int n0 = blockIdx.x * BN;
  const int wm = (wid >> 2) * 32;   // 2 wave-rows
  const int wn = (wid & 3) * 32;    // 4 wave-cols

  v8f acc[2][2];
  const v8f zf = {0.f, 0.f, 0.f, 0.f, 0.f, 0.f, 0.f, 0.f};
#pragma unroll
  for (int i = 0; i < 2; ++i)
#pragma unroll
    for (int j = 0; j < 2; ++j) acc[i][j] = zf;

  float4 ra[4];   // A tile in regs: 64x64 / 256thr / 4 = 4 quads/thread
  float4 rb[8];   // W tile in regs: 128x64 / 256thr / 4 = 8 quads/thread

  auto loadA = [&](int k0) {
#pragma unroll
    for (int it = 0; it < 4; ++it) {
      int q = it * 256 + tid;
      int m = q >> 4;              // 16 quads per 64-float row
      int kk = (q & 15) * 4;
      ra[it] = *(const float4*)&A[(size_t)(m0 + m) * K + k0 + kk];
    }
  };
  auto loadB = [&](int k0) {
    if (wTrans) {                  // W is (N,K): rows are K-contiguous
#pragma unroll
      for (int it = 0; it < 8; ++it) {
        int q = it * 256 + tid;
        int n = q >> 4;
        int kk = (q & 15) * 4;
        rb[it] = *(const float4*)&W[(size_t)(n0 + n) * K + k0 + kk];
      }
    } else {                       // W is (K,N): read along N (coalesced)
#pragma unroll
      for (int it = 0; it < 8; ++it) {
        int q = it * 256 + tid;
        int kk = q >> 5;           // 32 quads per 128-float row
        int n = (q & 31) * 4;
        rb[it] = *(const float4*)&W[(size_t)(k0 + kk) * N + n0 + n];
      }
    }
  };
  auto storeA = [&]() {
#pragma unroll
    for (int it = 0; it < 4; ++it) {
      int q = it * 256 + tid;
      int m = q >> 4;
      int kk = (q & 15) * 4;
      store_bf16x4(&lA[m][kk], ra[it]);
    }
  };
  auto storeB = [&]() {
    if (wTrans) {
#pragma unroll
      for (int it = 0; it < 8; ++it) {
        int q = it * 256 + tid;
        int n = q >> 4;
        int kk = (q & 15) * 4;
        store_bf16x4(&lB[n][kk], rb[it]);
      }
    } else {                       // transpose during LDS store
#pragma unroll
      for (int it = 0; it < 8; ++it) {
        int q = it * 256 + tid;
        int kk = q >> 5;
        int n = (q & 31) * 4;
        lB[n + 0][kk] = f2bf(rb[it].x);
        lB[n + 1][kk] = f2bf(rb[it].y);
        lB[n + 2][kk] = f2bf(rb[it].z);
        lB[n + 3][kk] = f2bf(rb[it].w);
      }
    }
  };

  loadA(0);
  loadB(0);
  for (int k0 = 0; k0 < K; k0 += BK) {
    storeA();
    storeB();
    __syncthreads();
    int kn = k0 + BK;
    if (kn < K) {        // issue next tile's loads; consumed by next storeA/B
      loadA(kn);
      loadB(kn);
    }
#pragma unroll
    for (int ks = 0; ks < BK / 32; ++ks) {
      v16bf af[2], bfg[2];
#pragma unroll
      for (int im = 0; im < 2; ++im)
        af[im] = load_a_frag(&lA[wm + im * 16 + l16][ks * 32], half);
#pragma unroll
      for (int in = 0; in < 2; ++in)
        bfg[in] = load_b_frag(&lB[wn + in * 16 + l16][ks * 32], half);
#pragma unroll
      for (int im = 0; im < 2; ++im)
#pragma unroll
        for (int in = 0; in < 2; ++in)
          acc[im][in] = wmma_bf16(af[im], bfg[in], acc[im][in]);
    }
    __syncthreads();
  }
  // epilogue; C/D layout: row = r + 8*half, col = l16
#pragma unroll
  for (int im = 0; im < 2; ++im)
#pragma unroll
    for (int in = 0; in < 2; ++in)
#pragma unroll
      for (int r = 0; r < 8; ++r) {
        int gm = m0 + wm + im * 16 + r + 8 * half;
        int gn = n0 + wn + in * 16 + l16;
        float v = acc[im][in][r];
        if (rowScale) v *= rowScale[(size_t)gm * rsStride];
        size_t offc = (size_t)gm * N + gn;
        if (residual) v += residual[offc];
        if (accumulate) v += C[offc];
        C[offc] = v;
      }
}

// ================== flash attention (WMMA) ==================
// grid: (B*NH, S/64), block 128 (4 waves x 16 q-rows)
__global__ __launch_bounds__(128)
void attn_kernel(const float* __restrict__ xq,    // [T, NH, 2*HD] (q normed+roped)
                 const float* __restrict__ cK,    // [B,NKV,MC,HD]
                 const float* __restrict__ cV,
                 float* __restrict__ attn_out) {  // [T, NH*HD]
  __shared__ __attribute__((aligned(16))) bf16_t lK[32][HD];   // [key][hd]
  __shared__ __attribute__((aligned(16))) bf16_t lVT[HD][32];  // [hd][key]
  __shared__ __attribute__((aligned(16))) bf16_t lP[4][16][32];
  const int bh = blockIdx.x;
  const int b = bh / NH, h = bh % NH;
  const int kv = h / (NH / NKV);
  const int q0 = blockIdx.y * 64;
  const int tid = threadIdx.x, lane = tid & 31, wid = tid >> 5;
  const int half = lane >> 4, l16 = lane & 15;
  const int qw = q0 + wid * 16;
  const float sm_scale = 0.08838834764831845f;  // 1/sqrt(128)

  // Q fragments (A layout), 4 chunks of K=32 over HD=128; float4 loads
  v16bf qf[4];
  {
    const float* qrow = xq + (((size_t)(b * SSQ + qw + l16)) * NH + h) * (2 * HD);
    float4 fq[16];
#pragma unroll
    for (int c = 0; c < 4; ++c) {
      fq[c * 4 + 0] = *(const float4*)&qrow[c * 32 + half * 8];
      fq[c * 4 + 1] = *(const float4*)&qrow[c * 32 + half * 8 + 4];
      fq[c * 4 + 2] = *(const float4*)&qrow[c * 32 + 16 + half * 8];
      fq[c * 4 + 3] = *(const float4*)&qrow[c * 32 + 16 + half * 8 + 4];
    }
#pragma unroll
    for (int c = 0; c < 4; ++c) {
      union { v16bf v; bf16_t e[16]; } u;
#pragma unroll
      for (int j = 0; j < 4; ++j) {
        float4 f = fq[c * 4 + j];
        u.e[j * 4 + 0] = f2bf(f.x); u.e[j * 4 + 1] = f2bf(f.y);
        u.e[j * 4 + 2] = f2bf(f.z); u.e[j * 4 + 3] = f2bf(f.w);
      }
      qf[c] = u.v;
    }
  }
  v8f o[8];
  const v8f zf = {0.f, 0.f, 0.f, 0.f, 0.f, 0.f, 0.f, 0.f};
#pragma unroll
  for (int f = 0; f < 8; ++f) o[f] = zf;
  float mrow[8], lrow[8];
#pragma unroll
  for (int r = 0; r < 8; ++r) { mrow[r] = -3.0e38f; lrow[r] = 0.f; }

  const float* Kbase = cK + ((size_t)(b * NKV + kv)) * MCC * HD;
  const float* Vbase = cV + ((size_t)(b * NKV + kv)) * MCC * HD;
  const int kend = q0 + 64;  // causal: k <= q, max q in this WG is q0+63

  for (int k0 = 0; k0 < kend; k0 += 32) {
    // stage K (row copy) + V (transposed): batch all 16 b128 loads, wait once
    {
      float4 fk[8], fv[8];
#pragma unroll
      for (int it = 0; it < 8; ++it) {
        int q = it * 128 + tid;
        int kk = q >> 5;          // 32 quads per 128-float row
        int hd = (q & 31) * 4;
        fk[it] = *(const float4*)&Kbase[(size_t)(k0 + kk) * HD + hd];
        fv[it] = *(const float4*)&Vbase[(size_t)(k0 + kk) * HD + hd];
      }
#pragma unroll
      for (int it = 0; it < 8; ++it) {
        int q = it * 128 + tid;
        int kk = q >> 5;
        int hd = (q & 31) * 4;
        store_bf16x4(&lK[kk][hd], fk[it]);
        lVT[hd + 0][kk] = f2bf(fv[it].x);
        lVT[hd + 1][kk] = f2bf(fv[it].y);
        lVT[hd + 2][kk] = f2bf(fv[it].z);
        lVT[hd + 3][kk] = f2bf(fv[it].w);
      }
    }
    __syncthreads();

    // S = Q @ K^T  (two 16-col fragments)
    v8f sfr[2]; sfr[0] = zf; sfr[1] = zf;
#pragma unroll
    for (int cc = 0; cc < 2; ++cc)
#pragma unroll
      for (int c = 0; c < 4; ++c)
        sfr[cc] = wmma_bf16(qf[c], load_b_frag(&lK[cc * 16 + l16][c * 32], half),
                            sfr[cc]);

    // scale + causal mask + chunk row-max (reduction across 16-lane half)
    float mc[8];
#pragma unroll
    for (int r = 0; r < 8; ++r) {
      int ql = qw + r + 8 * half;
      float s0 = sfr[0][r] * sm_scale;
      float s1 = sfr[1][r] * sm_scale;
      if (k0 + l16 > ql)      s0 = -3.0e38f;
      if (k0 + 16 + l16 > ql) s1 = -3.0e38f;
      sfr[0][r] = s0; sfr[1][r] = s1;
      float mx = fmaxf(s0, s1);
#pragma unroll
      for (int d = 1; d < 16; d <<= 1) mx = fmaxf(mx, __shfl_xor(mx, d, 32));
      mc[r] = mx;
    }
    // online softmax update + write P to per-wave LDS (C->A relayout)
#pragma unroll
    for (int r = 0; r < 8; ++r) {
      float mnew = fmaxf(mrow[r], mc[r]);
      float alpha = __expf(mrow[r] - mnew);
      mrow[r] = mnew;
      float p0 = __expf(sfr[0][r] - mnew);
      float p1 = __expf(sfr[1][r] - mnew);
      float rs = p0 + p1;
#pragma unroll
      for (int d = 1; d < 16; d <<= 1) rs += __shfl_xor(rs, d, 32);
      lrow[r] = lrow[r] * alpha + rs;
#pragma unroll
      for (int f = 0; f < 8; ++f) o[f][r] *= alpha;
      lP[wid][r + 8 * half][l16] = f2bf(p0);
      lP[wid][r + 8 * half][16 + l16] = f2bf(p1);
    }
    // O += P @ V
    v16bf pa = load_a_frag(&lP[wid][l16][0], half);
#pragma unroll
    for (int f = 0; f < 8; ++f)
      o[f] = wmma_bf16(pa, load_b_frag(&lVT[f * 16 + l16][0], half), o[f]);
    __syncthreads();
  }
  // epilogue: normalize, sigmoid gate from xq[..., HD:], store
#pragma unroll
  for (int r = 0; r < 8; ++r) {
    int ql = qw + r + 8 * half;
    size_t t = (size_t)b * SSQ + ql;
    float inv = 1.0f / lrow[r];
#pragma unroll
    for (int f = 0; f < 8; ++f) {
      int d = f * 16 + l16;
      float g = xq[(t * NH + h) * (2 * HD) + HD + d];
      attn_out[t * (NH * HD) + h * HD + d] = o[f][r] * inv * sigmoidf_(g);
    }
  }
}

// ================== elementwise / small kernels ==================
__global__ __launch_bounds__(256)
void rmsnorm_kernel(const float* __restrict__ x, const float* __restrict__ w,
                    float* __restrict__ out) {
  int t = blockIdx.x, tid = threadIdx.x;
  __shared__ float red[256];
  const float* row = x + (size_t)t * DIMD;
  float s = 0.f;
#pragma unroll
  for (int it = 0; it < DIMD / (256 * 4); ++it) {  // 2 iters, float4
    float4 f = *(const float4*)&row[(it * 256 + tid) * 4];
    s += f.x * f.x + f.y * f.y + f.z * f.z + f.w * f.w;
  }
  red[tid] = s; __syncthreads();
  for (int st = 128; st > 0; st >>= 1) {
    if (tid < st) red[tid] += red[tid + st];
    __syncthreads();
  }
  float r = rsqrtf(red[0] * (1.0f / DIMD) + EPSF);
  float* orow = out + (size_t)t * DIMD;
#pragma unroll
  for (int it = 0; it < DIMD / (256 * 4); ++it) {
    int d = (it * 256 + tid) * 4;
    float4 f = *(const float4*)&row[d];
    float4 wv = *(const float4*)&w[d];
    float4 o;
    o.x = f.x * r * (1.0f + wv.x); o.y = f.y * r * (1.0f + wv.y);
    o.z = f.z * r * (1.0f + wv.z); o.w = f.w * r * (1.0f + wv.w);
    *(float4*)&orow[d] = o;
  }
}

// per-head rms-norm + RoPE, in place on q part of xq
__global__ __launch_bounds__(HD)
void q_post_kernel(float* __restrict__ xq, const float* __restrict__ qnorm,
                   const float* __restrict__ freqs, const int* __restrict__ seqpos) {
  int idx = blockIdx.x;
  int h = idx % NH, s = (idx / NH) % SSQ, b = idx / (NH * SSQ);
  int d = threadIdx.x;
  __shared__ float red[HD];
  __shared__ float nb[HD];
  size_t xoff = (((size_t)(b * SSQ + s)) * NH + h) * (2 * HD) + d;
  float v = xq[xoff];
  red[d] = v * v; __syncthreads();
  for (int st = HD / 2; st > 0; st >>= 1) {
    if (d < st) red[d] += red[d + st];
    __syncthreads();
  }
  float nv = v * rsqrtf(red[0] * (1.0f / HD) + EPSF) * (1.0f + qnorm[d]);
  nb[d] = nv; __syncthreads();
  int pos = seqpos[b] + s;
  float outv;
  if (d < ROT) {
    int p = d >> 1;
    float c  = freqs[((size_t)pos * (ROT / 2) + p) * 2 + 0];
    float sn = freqs[((size_t)pos * (ROT / 2) + p) * 2 + 1];
    float x0 = nb[2 * p], x1 = nb[2 * p + 1];
    outv = (d & 1) ? (x0 * sn + x1 * c) : (x0 * c - x1 * sn);
  } else outv = nv;
  xq[xoff] = outv;
}

// per-kv-head rms-norm + RoPE on K, write K and V into output caches
__global__ __launch_bounds__(HD)
void kv_post_kernel(const float* __restrict__ xk, const float* __restrict__ xv,
                    const float* __restrict__ knorm, const float* __restrict__ freqs,
                    const int* __restrict__ seqpos,
                    float* __restrict__ cK, float* __restrict__ cV) {
  int idx = blockIdx.x;
  int kvh = idx % NKV, s = (idx / NKV) % SSQ, b = idx / (NKV * SSQ);
  int d = threadIdx.x;
  __shared__ float red[HD];
  __shared__ float nb[HD];
  size_t xoff = (((size_t)(b * SSQ + s)) * NKV + kvh) * HD + d;
  float v = xk[xoff];
  red[d] = v * v; __syncthreads();
  for (int st = HD / 2; st > 0; st >>= 1) {
    if (d < st) red[d] += red[d + st];
    __syncthreads();
  }
  float nv = v * rsqrtf(red[0] * (1.0f / HD) + EPSF) * (1.0f + knorm[d]);
  nb[d] = nv; __syncthreads();
  int pos = seqpos[b] + s;
  float outv;
  if (d < ROT) {
    int p = d >> 1;
    float c  = freqs[((size_t)pos * (ROT / 2) + p) * 2 + 0];
    float sn = freqs[((size_t)pos * (ROT / 2) + p) * 2 + 1];
    float x0 = nb[2 * p], x1 = nb[2 * p + 1];
    outv = (d & 1) ? (x0 * sn + x1 * c) : (x0 * c - x1 * sn);
  } else outv = nv;
  size_t coff = (((size_t)b * NKV + kvh) * MCC + pos) * HD + d;
  cK[coff] = outv;
  cV[coff] = xv[xoff];
}

__global__ __launch_bounds__(256)
void router_kernel(const float* __restrict__ hf, const float* __restrict__ rw,
                   float* __restrict__ wout) {
  int t = blockIdx.x, tid = threadIdx.x, lane = tid & 31, wid = tid >> 5;
  float part[NE];
#pragma unroll
  for (int e = 0; e < NE; ++e) part[e] = 0.f;
  const float* row = hf + (size_t)t * DIMD;
  for (int d = tid; d < DIMD; d += 256) {
    float hv = row[d];
#pragma unroll
    for (int e = 0; e < NE; ++e) part[e] += hv * rw[e * DIMD + d];
  }
  __shared__ float red[8][NE];
#pragma unroll
  for (int e = 0; e < NE; ++e) {
    float v = part[e];
#pragma unroll
    for (int d2 = 1; d2 < 32; d2 <<= 1) v += __shfl_xor(v, d2, 32);
    if (lane == 0) red[wid][e] = v;
  }
  __syncthreads();
  if (tid == 0) {
    float pr[NE];
    float mx = -3.0e38f;
    for (int e = 0; e < NE; ++e) {
      float v = 0.f;
      for (int wv = 0; wv < 8; ++wv) v += red[wv][e];
      pr[e] = v;
      mx = fmaxf(mx, v);
    }
    float se = 0.f;
    for (int e = 0; e < NE; ++e) { pr[e] = __expf(pr[e] - mx); se += pr[e]; }
    for (int e = 0; e < NE; ++e) pr[e] /= se;
    int i0 = 0;
    for (int e = 1; e < NE; ++e) if (pr[e] > pr[i0]) i0 = e;
    int i1 = (i0 == 0) ? 1 : 0;
    for (int e = 0; e < NE; ++e) if (e != i0 && pr[e] > pr[i1]) i1 = e;
    float sum2 = pr[i0] + pr[i1];
    for (int e = 0; e < NE; ++e) wout[(size_t)t * NE + e] = 0.f;
    wout[(size_t)t * NE + i0] = pr[i0] / sum2;
    wout[(size_t)t * NE + i1] = pr[i1] / sum2;
  }
}

__global__ __launch_bounds__(256)
void sgate_kernel(const float* __restrict__ hf, const float* __restrict__ seg,
                  float* __restrict__ sg) {
  int t = blockIdx.x, tid = threadIdx.x;
  __shared__ float red[256];
  const float* row = hf + (size_t)t * DIMD;
  float s = 0.f;
  for (int d = tid; d < DIMD; d += 256) s += row[d] * seg[d];
  red[tid] = s; __syncthreads();
  for (int st = 128; st > 0; st >>= 1) {
    if (tid < st) red[tid] += red[tid + st];
    __syncthreads();
  }
  if (tid == 0) sg[t] = sigmoidf_(red[0]);
}

__global__ void silu_mul_expert_kernel(const float* __restrict__ gu,
                                       float* __restrict__ act) {
  int i = blockIdx.x * blockDim.x + threadIdx.x;
  if (i >= TT * MIH) return;
  int t = i / MIH, f = i % MIH;
  float g = gu[(size_t)t * (2 * MIH) + f];
  float u = gu[(size_t)t * (2 * MIH) + MIH + f];
  act[(size_t)t * MIH + f] = g * sigmoidf_(g) * u;
}

__global__ void silu_mul_shared_kernel(float* __restrict__ g,
                                       const float* __restrict__ u) {
  size_t i = (size_t)blockIdx.x * blockDim.x + threadIdx.x;
  if (i >= (size_t)TT * SIH) return;
  float gv = g[i];
  g[i] = gv * sigmoidf_(gv) * u[i];
}

__global__ void add_kernel(const float* __restrict__ a, const float* __restrict__ b,
                           float* __restrict__ out, size_t n) {
  size_t i = (size_t)blockIdx.x * blockDim.x + threadIdx.x;
  if (i < n) out[i] = a[i] + b[i];
}

__global__ void zero_kernel(float* __restrict__ p, size_t n) {
  size_t i = (size_t)blockIdx.x * blockDim.x + threadIdx.x;
  if (i < n) p[i] = 0.f;
}

// ================== launch ==================
extern "C" void kernel_launch(void* const* d_in, const int* in_sizes, int n_in,
                              void* d_out, int out_size, void* d_ws, size_t ws_size,
                              hipStream_t stream) {
  (void)in_sizes; (void)n_in; (void)out_size; (void)ws_size;
  const float* x       = (const float*)d_in[0];
  const float* freqs   = (const float*)d_in[1];
  const float* cK_in   = (const float*)d_in[2];
  const float* cV_in   = (const float*)d_in[3];
  const int*   seqpos  = (const int*)d_in[4];
  /* d_in[5] = mask: causal, computed analytically in attn_kernel */
  const float* wq      = (const float*)d_in[6];
  const float* wk      = (const float*)d_in[7];
  const float* wv      = (const float*)d_in[8];
  const float* wo      = (const float*)d_in[9];
  const float* q_norm  = (const float*)d_in[10];
  const float* k_norm  = (const float*)d_in[11];
  const float* attn_nw = (const float*)d_in[12];
  const float* ffn_nw  = (const float*)d_in[13];
  const float* rw      = (const float*)d_in[14];
  const float* egu     = (const float*)d_in[15];
  const float* edw     = (const float*)d_in[16];
  const float* sh_g    = (const float*)d_in[17];
  const float* sh_u    = (const float*)d_in[18];
  const float* sh_d    = (const float*)d_in[19];
  const float* sh_eg   = (const float*)d_in[20];

  float* out    = (float*)d_out;
  float* cK_out = out + (size_t)TT * DIMD;
  float* cV_out = cK_out + (size_t)BB * NKV * MCC * HD;

  float* ws = (float*)d_ws;
  size_t off = 0;
  auto alloc = [&](size_t n) { float* p = ws + off; off += n; return p; };
  float* h    = alloc((size_t)TT * DIMD);
  float* xqb  = alloc((size_t)TT * NH * 2 * HD);
  float* xkb  = alloc((size_t)TT * NKV * HD);
  float* xvb  = alloc((size_t)TT * NKV * HD);
  float* attn = alloc((size_t)TT * NH * HD);
  float* ares = alloc((size_t)TT * DIMD);
  float* hf   = alloc((size_t)TT * DIMD);
  float* wbuf = alloc((size_t)TT * NE);
  float* sg   = alloc((size_t)TT);
  float* gu   = alloc((size_t)TT * 2 * MIH);
  float* act  = alloc((size_t)TT * MIH);
  float* moe  = alloc((size_t)TT * DIMD);
  float* gbuf = alloc((size_t)TT * SIH);
  float* ubuf = alloc((size_t)TT * SIH);

  auto gemm = [&](const float* A, const float* W, float* C, int M, int N, int K,
                  int wTrans, const float* resid, const float* rs, int rss,
                  int accum) {
    dim3 grid(N / BN, M / BM);
    gemm_wmma<<<grid, 256, 0, stream>>>(A, W, C, M, N, K, wTrans, resid, rs, rss,
                                        accum);
  };

  // caches: copy input -> output, then overwrite rows [pos, pos+S)
  size_t cbytes = sizeof(float) * (size_t)BB * NKV * MCC * HD;
  hipMemcpyAsync(cK_out, cK_in, cbytes, hipMemcpyDeviceToDevice, stream);
  hipMemcpyAsync(cV_out, cV_in, cbytes, hipMemcpyDeviceToDevice, stream);

  // attention path
  rmsnorm_kernel<<<TT, 256, 0, stream>>>(x, attn_nw, h);
  gemm(h, wq, xqb, TT, NH * 2 * HD, DIMD, 0, nullptr, nullptr, 0, 0);
  gemm(h, wk, xkb, TT, NKV * HD, DIMD, 0, nullptr, nullptr, 0, 0);
  gemm(h, wv, xvb, TT, NKV * HD, DIMD, 0, nullptr, nullptr, 0, 0);
  q_post_kernel<<<BB * SSQ * NH, HD, 0, stream>>>(xqb, q_norm, freqs, seqpos);
  kv_post_kernel<<<BB * SSQ * NKV, HD, 0, stream>>>(xkb, xvb, k_norm, freqs,
                                                    seqpos, cK_out, cV_out);
  {
    dim3 agrid(BB * NH, SSQ / 64);
    attn_kernel<<<agrid, 128, 0, stream>>>(xqb, cK_out, cV_out, attn);
  }
  gemm(attn, wo, ares, TT, DIMD, NH * HD, 0, x, nullptr, 0, 0);  // + residual x

  // MoE path
  rmsnorm_kernel<<<TT, 256, 0, stream>>>(ares, ffn_nw, hf);
  router_kernel<<<TT, 256, 0, stream>>>(hf, rw, wbuf);
  sgate_kernel<<<TT, 256, 0, stream>>>(hf, sh_eg, sg);
  {
    size_t n = (size_t)TT * DIMD;
    zero_kernel<<<(int)((n + 255) / 256), 256, 0, stream>>>(moe, n);
  }
  for (int e = 0; e < NE; ++e) {
    gemm(hf, egu + (size_t)e * 2 * MIH * DIMD, gu, TT, 2 * MIH, DIMD, 1, nullptr,
         nullptr, 0, 0);
    int n1 = TT * MIH;
    silu_mul_expert_kernel<<<(n1 + 255) / 256, 256, 0, stream>>>(gu, act);
    gemm(act, edw + (size_t)e * DIMD * MIH, moe, TT, DIMD, MIH, 1, nullptr,
         wbuf + e, NE, 1);  // moe += down * w[:,e]
  }
  // shared expert
  gemm(hf, sh_g, gbuf, TT, SIH, DIMD, 0, nullptr, nullptr, 0, 0);
  gemm(hf, sh_u, ubuf, TT, SIH, DIMD, 0, nullptr, nullptr, 0, 0);
  {
    size_t n2 = (size_t)TT * SIH;
    silu_mul_shared_kernel<<<(int)((n2 + 255) / 256), 256, 0, stream>>>(gbuf, ubuf);
  }
  gemm(gbuf, sh_d, moe, TT, DIMD, SIH, 0, nullptr, sg, 1, 1);  // moe += sh * sg

  // out = (x + attn@wo) + (moe + shared)
  size_t n3 = (size_t)TT * DIMD;
  add_kernel<<<(int)((n3 + 255) / 256), 256, 0, stream>>>(ares, moe, out, n3);
}
